// BiMamba2Model_28166395527507
// MI455X (gfx1250) — compile-verified, matrix operand order checked by reading, not run on hardware
//
#include <hip/hip_runtime.h>
#include <cstdint>
#include <cstddef>

// ---------------------------------------------------------------------------
// BiMamba2 forward for MI455X (gfx1250): all GEMM-shaped math on
// v_wmma_f32_16x16x32_bf16 (bf16 operands, f32 accumulate), elementwise in f32.
// wave32 everywhere; WMMA only under wave-uniform control flow.
// GEMM launch contract: M % 128 == 0, K % 32 == 0 (all call sites satisfy it).
// ---------------------------------------------------------------------------

typedef __bf16 bf16;
typedef bf16  bfx16 __attribute__((ext_vector_type(16)));
typedef float fx8   __attribute__((ext_vector_type(8)));

// ---- model dims -----------------------------------------------------------
constexpr int BSZ    = 4;
constexpr int SEQ    = 2048;
constexpr int DMODEL = 512;
constexpr int DINNER = 1024;
constexpr int NH     = 16;     // heads (SSD)
constexpr int HD     = 64;     // headdim
constexpr int DST    = 64;     // dstate
constexpr int CHK    = 256;    // chunk
constexpr int NCH    = 8;      // chunks per seq
constexpr int CONVD  = 1152;
constexpr int DINPRJ = 2192;
constexpr int MROWS  = BSZ * SEQ;            // 8192
constexpr int NQR    = BSZ * (SEQ / 2);      // 4096 query rows
constexpr int CAH    = 4;
constexpr int CAHD   = 128;
constexpr int NHID   = 1024;
constexpr int NOUT   = 10;

// ---- workspace layout (float offsets) -------------------------------------
constexpr size_t OH      = 0;
constexpr size_t ORES    = OH      + (size_t)MROWS * DMODEL;
constexpr size_t OHN     = ORES    + (size_t)MROWS * DMODEL;
constexpr size_t OFLIP   = OHN     + (size_t)MROWS * DMODEL;
constexpr size_t OZ      = OFLIP   + (size_t)MROWS * DMODEL;
constexpr size_t OXBC    = OZ      + (size_t)MROWS * DINNER;
constexpr size_t OXC     = OXBC    + (size_t)MROWS * CONVD;
constexpr size_t OY      = OXC     + (size_t)MROWS * CONVD;
constexpr size_t ODIR    = OY      + (size_t)MROWS * DINNER;
constexpr size_t OST     = ODIR    + (size_t)MROWS * DMODEL;
constexpr size_t ODT     = OST     + (size_t)BSZ * NCH * NH * HD * DST;
constexpr size_t OACUM   = ODT     + (size_t)MROWS * NH;
constexpr size_t OCSUM   = OACUM   + (size_t)MROWS * NH;
constexpr size_t ODTRAW  = OCSUM   + (size_t)BSZ * NCH * NH;

// attention-stage aliases (regions free after the mamba layers)
constexpr size_t OQRAW = OZ;
constexpr size_t OCRAW = OZ + (size_t)NQR * DMODEL;
constexpr size_t OQIN  = OZ + 2 * (size_t)NQR * DMODEL;
constexpr size_t OCIN  = OZ + 3 * (size_t)NQR * DMODEL;
constexpr size_t OQN   = OFLIP;
constexpr size_t OCN   = OFLIP + (size_t)NQR * DMODEL;
constexpr size_t OQM   = OXBC;
constexpr size_t OKM   = OXBC + (size_t)NQR * DMODEL;
constexpr size_t OVM   = OXBC + 2 * (size_t)NQR * DMODEL;
constexpr size_t OOM   = OXBC + 3 * (size_t)NQR * DMODEL;
constexpr size_t OS    = OXC;            // 16.78M floats, spans XC+Y regions
constexpr size_t OATT  = OH;
constexpr size_t OD1   = ODIR;

// ---- small device helpers -------------------------------------------------
__device__ __forceinline__ bf16 tobf(float x) { return (bf16)x; }
__device__ __forceinline__ float fbf(bf16 x) { return (float)x; }
__device__ __forceinline__ unsigned pk2(float lo, float hi) {
  bf16 l = (bf16)lo, h = (bf16)hi;
  unsigned short ul, uh;
  __builtin_memcpy(&ul, &l, 2); __builtin_memcpy(&uh, &h, 2);
  return (unsigned)ul | ((unsigned)uh << 16);
}
__device__ __forceinline__ uint4 pk8(float4 a, float4 b) {
  uint4 r; r.x = pk2(a.x, a.y); r.y = pk2(a.z, a.w);
  r.z = pk2(b.x, b.y); r.w = pk2(b.z, b.w); return r;
}
__device__ __forceinline__ uint4 pk8s(float4 a, float4 b, float s) {
  uint4 r; r.x = pk2(a.x * s, a.y * s); r.y = pk2(a.z * s, a.w * s);
  r.z = pk2(b.x * s, b.y * s); r.w = pk2(b.z * s, b.w * s); return r;
}
__device__ __forceinline__ float siluf(float x) { return x / (1.f + __expf(-x)); }
__device__ __forceinline__ fx8 wmma_bf16(bfx16 a, bfx16 b, fx8 c) {
  return __builtin_amdgcn_wmma_f32_16x16x32_bf16(false, a, false, b, (short)0, c,
                                                 false, false);
}

// ===========================================================================
// Generic GEMM: C[M,N] (+)= A[M,K] * W(opW? W[N,K]^T : W[K,N]) + bias
// bf16 WMMA, f32 accumulate. 256 threads = 8 waves; 128x128 block tile,
// BK=32, each wave computes 32x64 (2 A-frags x 4 B-frags -> 8 WMMA / step).
// Contract: M % 128 == 0, K % 32 == 0; N arbitrary.
// ===========================================================================
__global__ void __launch_bounds__(256) gemm_bf16_kernel(
    const float* __restrict__ A, int lda,
    const float* __restrict__ W, int ldw, int opW,
    const float* __restrict__ bias,
    float* __restrict__ C, int ldc,
    int M, int N, int K, int accum)
{
  __shared__ __align__(16) bf16 As[128][32];
  __shared__ __align__(16) bf16 Bs[128][32];   // stored as B^T: [n][k]
  const int row0 = blockIdx.y * 128;
  const int col0 = blockIdx.x * 128;
  const int tid  = threadIdx.x;
  const int wv = tid >> 5, lane = tid & 31, hx = (lane >> 4) & 1, lr = lane & 15;
  const int g0 = hx * 8, g1 = 16 + hx * 8;
  const int wrow = (wv & 3) * 32, wcol = (wv >> 2) * 64;
  const bool fullN = (col0 + 128) <= N;

  fx8 acc[2][4];
  for (int a = 0; a < 2; ++a)
    for (int i = 0; i < 4; ++i)
      for (int e = 0; e < 8; ++e) acc[a][i][e] = 0.f;

  for (int k0 = 0; k0 < K; k0 += 32) {
    { // A tile 128x32: vectorized, no guards (M%128==0, K%32==0)
      int r = tid >> 1, kb = (tid & 1) * 16;
      const float* ap = A + (size_t)(row0 + r) * lda + k0 + kb;
      const float4* av = (const float4*)ap;
      float4 f0 = av[0], f1 = av[1], f2 = av[2], f3 = av[3];
      *(uint4*)&As[r][kb]     = pk8(f0, f1);
      *(uint4*)&As[r][kb + 8] = pk8(f2, f3);
      if (k0 + 32 < K) __builtin_prefetch(ap + 32);
    }
    if (opW) { // W[N,K]: contiguous along k
      int n = tid >> 1, kb = (tid & 1) * 16;
      if (col0 + n < N) {
        const float4* wv4 = (const float4*)(W + (size_t)(col0 + n) * ldw + k0 + kb);
        float4 f0 = wv4[0], f1 = wv4[1], f2 = wv4[2], f3 = wv4[3];
        *(uint4*)&Bs[n][kb]     = pk8(f0, f1);
        *(uint4*)&Bs[n][kb + 8] = pk8(f2, f3);
      } else {
        uint4 z; z.x = z.y = z.z = z.w = 0u;
        *(uint4*)&Bs[n][kb] = z; *(uint4*)&Bs[n][kb + 8] = z;
      }
    } else {   // W[K,N]: contiguous along n, transpose into Bs
      int k = tid & 31, nb = (tid >> 5) * 16;
      if (fullN) {
        const float4* wv4 = (const float4*)(W + (size_t)(k0 + k) * ldw + col0 + nb);
        float4 f0 = wv4[0], f1 = wv4[1], f2 = wv4[2], f3 = wv4[3];
        float fv[16] = {f0.x,f0.y,f0.z,f0.w, f1.x,f1.y,f1.z,f1.w,
                        f2.x,f2.y,f2.z,f2.w, f3.x,f3.y,f3.z,f3.w};
        #pragma unroll
        for (int j = 0; j < 16; ++j) Bs[nb + j][k] = tobf(fv[j]);
      } else {
        #pragma unroll
        for (int j = 0; j < 16; ++j) {
          int col = col0 + nb + j;
          Bs[nb + j][k] = tobf(col < N ? W[(size_t)(k0 + k) * ldw + col] : 0.f);
        }
      }
    }
    __syncthreads();
    bfx16 af[2];
    #pragma unroll
    for (int ar = 0; ar < 2; ++ar) {
      int arow = wrow + ar * 16 + lr;
      #pragma unroll
      for (int j = 0; j < 8; ++j) {
        af[ar][j] = As[arow][g0 + j]; af[ar][8 + j] = As[arow][g1 + j];
      }
    }
    #pragma unroll
    for (int cb = 0; cb < 4; ++cb) {
      int bcol = wcol + cb * 16 + lr;
      bfx16 bfv;
      #pragma unroll
      for (int j = 0; j < 8; ++j) { bfv[j] = Bs[bcol][g0 + j]; bfv[8 + j] = Bs[bcol][g1 + j]; }
      acc[0][cb] = wmma_bf16(af[0], bfv, acc[0][cb]);
      acc[1][cb] = wmma_bf16(af[1], bfv, acc[1][cb]);
    }
    __syncthreads();
  }
  #pragma unroll
  for (int ar = 0; ar < 2; ++ar) {
    #pragma unroll
    for (int cb = 0; cb < 4; ++cb) {
      int col = col0 + wcol + cb * 16 + lr;
      if (col >= N) continue;
      float bv = bias ? bias[col] : 0.f;
      #pragma unroll
      for (int r = 0; r < 8; ++r) {
        int row = row0 + wrow + ar * 16 + r + hx * 8;
        size_t idx = (size_t)row * ldc + col;
        float v = acc[ar][cb][r] + bv;
        if (accum) v += C[idx];
        C[idx] = v;
      }
    }
  }
}

// ===========================================================================
// LayerNorm over D=512 rows
// ===========================================================================
__global__ void __launch_bounds__(256) ln_kernel(
    const float* __restrict__ x, const float* __restrict__ w,
    const float* __restrict__ bb, float* __restrict__ out, int D)
{
  __shared__ float buf[512];
  __shared__ float red[256];
  int row = blockIdx.x, t = threadIdx.x;
  const float* xr = x + (size_t)row * D;
  float s = 0.f;
  for (int c = t; c < D; c += 256) { float v = xr[c]; buf[c] = v; s += v; }
  red[t] = s; __syncthreads();
  for (int o = 128; o > 0; o >>= 1) { if (t < o) red[t] += red[t + o]; __syncthreads(); }
  float mean = red[0] / D; __syncthreads();
  float vs = 0.f;
  for (int c = t; c < D; c += 256) { float d = buf[c] - mean; vs += d * d; }
  red[t] = vs; __syncthreads();
  for (int o = 128; o > 0; o >>= 1) { if (t < o) red[t] += red[t + o]; __syncthreads(); }
  float inv = rsqrtf(red[0] / D + 1e-5f);
  for (int c = t; c < D; c += 256)
    out[(size_t)row * D + c] = (buf[c] - mean) * inv * w[c] + bb[c];
}

// ===========================================================================
// dt = softplus(dtraw + dt_bias)      [MROWS x NH]
// ===========================================================================
__global__ void dt_kernel(const float* __restrict__ raw,
                          const float* __restrict__ dtb, float* __restrict__ dt)
{
  int idx = blockIdx.x * 256 + threadIdx.x;
  if (idx >= MROWS * NH) return;
  float x = raw[idx] + dtb[idx & (NH - 1)];
  dt[idx] = (x > 20.f) ? x : log1pf(__expf(x));
}

// ===========================================================================
// causal depthwise conv (CONV=4) + SiLU
// ===========================================================================
__global__ void conv_silu_kernel(const float* __restrict__ xbc,
                                 const float* __restrict__ cw,
                                 const float* __restrict__ cb,
                                 float* __restrict__ xc)
{
  size_t idx = (size_t)blockIdx.x * 256 + threadIdx.x;
  if (idx >= (size_t)MROWS * CONVD) return;
  int ch = (int)(idx % CONVD);
  size_t row = idx / CONVD;
  int tt = (int)(row & (SEQ - 1));
  size_t brow = row - tt;
  float acc = cb[ch];
  #pragma unroll
  for (int k = 0; k < 4; ++k) {
    int src = tt - 3 + k;
    if (src >= 0) acc += xbc[(brow + src) * CONVD + ch] * cw[ch * 4 + k];
  }
  xc[idx] = siluf(acc);
}

// ===========================================================================
// SSD part1: per (b,chunk,head) workgroup.
//   cumsum(A), Y_diag = (C B^T ∘ L) @ (x*dt), chunk state = (x*dt*dec)^T @ B
// all matmuls on bf16 WMMA, masks/exp in f32 on the accumulator.
// ===========================================================================
__global__ void __launch_bounds__(256) ssd_part1_kernel(
    const float* __restrict__ XCb, const float* __restrict__ DTb,
    const float* __restrict__ Alog,
    float* __restrict__ Yb, float* __restrict__ Stb,
    float* __restrict__ Acumb, float* __restrict__ Csumb)
{
  extern __shared__ char smem[];
  bf16*  sB   = (bf16*)smem;                     // [256][64] B
  bf16*  sC   = sB  + 256 * 64;                  // [256][64] C
  bf16*  sXd  = sC  + 256 * 64;                  // [256][64] x*dt
  float* sAc  = (float*)(smem + 3 * 256 * 64 * sizeof(bf16));
  float* sDec = sAc + 256;
  bf16*  Msc  = (bf16*)(sDec + 256);             // 8 waves x [16][36]

  int idx = blockIdx.x;
  int b = idx >> 7, c = (idx >> 4) & 7, hh = idx & 15;
  int t = threadIdx.x;
  size_t rowbase = (size_t)b * SEQ + (size_t)c * CHK;
  size_t grow = rowbase + t;
  const float* xr = XCb + grow * CONVD;
  float dtv = DTb[grow * NH + hh];
  #pragma unroll
  for (int n4 = 0; n4 < 64; n4 += 8) {
    const float4* pb = (const float4*)(xr + DINNER + n4);
    *(uint4*)&sB[t * 64 + n4] = pk8(pb[0], pb[1]);
    const float4* pc = (const float4*)(xr + DINNER + DST + n4);
    *(uint4*)&sC[t * 64 + n4] = pk8(pc[0], pc[1]);
    const float4* px = (const float4*)(xr + hh * HD + n4);
    *(uint4*)&sXd[t * 64 + n4] = pk8s(px[0], px[1], dtv);
  }
  sAc[t] = -__expf(Alog[hh]) * dtv;
  __syncthreads();
  // inclusive scan (Hillis-Steele)
  for (int off = 1; off < 256; off <<= 1) {
    float v = (t >= off) ? sAc[t - off] : 0.f;
    __syncthreads();
    sAc[t] += v;
    __syncthreads();
  }
  float csum = sAc[255];
  sDec[t] = __expf(csum - sAc[t]);
  Acumb[grow * NH + hh] = sAc[t];
  if (t == 0) Csumb[((size_t)b * NCH + c) * NH + hh] = csum;
  __syncthreads();

  int wv = t >> 5, lane = t & 31, hx = lane >> 4, lr = lane & 15;
  int g0 = hx * 8, g1 = 16 + hx * 8;
  bf16* myM = Msc + wv * (16 * 36);

  // ---- Y_diag: each wave owns 32 l-rows, all 64 p-cols --------------------
  for (int lt = 0; lt < 2; ++lt) {
    int l0 = wv * 32 + lt * 16;
    fx8 yacc[4];
    for (int pb = 0; pb < 4; ++pb)
      for (int e = 0; e < 8; ++e) yacc[pb][e] = 0.f;
    int ntiles = l0 / 16 + 1;                       // causal s-tiles
    for (int sb = 0; sb < ntiles; sb += 2) {
      for (int q = 0; q < 2; ++q) {                 // uniform per wave
        int st = sb + q;
        if (st < ntiles) {
          int s0 = st * 16;
          fx8 cbacc; for (int e = 0; e < 8; ++e) cbacc[e] = 0.f;
          for (int kk = 0; kk < 64; kk += 32) {     // K = n (dstate)
            bfx16 afr, bfr;
            int lrow = l0 + lr;
            #pragma unroll
            for (int j = 0; j < 8; ++j) {
              afr[j]     = sC[lrow * 64 + kk + g0 + j];
              afr[8 + j] = sC[lrow * 64 + kk + g1 + j];
            }
            int ss = s0 + lr;
            #pragma unroll
            for (int j = 0; j < 8; ++j) {
              bfr[j]     = sB[ss * 64 + kk + g0 + j];
              bfr[8 + j] = sB[ss * 64 + kk + g1 + j];
            }
            cbacc = wmma_bf16(afr, bfr, cbacc);
          }
          int scol = s0 + lr;
          #pragma unroll
          for (int r = 0; r < 8; ++r) {             // apply L mask + exp
            int lrow = l0 + r + hx * 8;
            float lv = (scol <= lrow) ? __expf(sAc[lrow] - sAc[scol]) : 0.f;
            myM[(lrow - l0) * 36 + q * 16 + lr] = tobf(cbacc[r] * lv);
          }
        } else {
          if (hx == 0)
            for (int r = 0; r < 16; ++r) myM[r * 36 + q * 16 + lr] = tobf(0.f);
        }
      }
      // M(16x32) @ xd(32x64)
      bfx16 afr;
      #pragma unroll
      for (int j = 0; j < 8; ++j) {
        afr[j]     = myM[lr * 36 + g0 + j];
        afr[8 + j] = myM[lr * 36 + g1 + j];
      }
      int sbase = sb * 16;
      #pragma unroll
      for (int pb = 0; pb < 4; ++pb) {
        bfx16 bfr;
        int p = pb * 16 + lr;
        #pragma unroll
        for (int j = 0; j < 8; ++j) {
          bfr[j]     = sXd[(sbase + g0 + j) * 64 + p];
          bfr[8 + j] = sXd[(sbase + g1 + j) * 64 + p];
        }
        yacc[pb] = wmma_bf16(afr, bfr, yacc[pb]);
      }
    }
    #pragma unroll
    for (int pb = 0; pb < 4; ++pb) {
      int p = pb * 16 + lr;
      #pragma unroll
      for (int r = 0; r < 8; ++r) {
        int lrow = l0 + r + hx * 8;
        Yb[(rowbase + lrow) * DINNER + (size_t)hh * HD + p] = yacc[pb][r];
      }
    }
  }

  // ---- chunk state[p,n] = sum_l xd[l,p]*dec[l]*B[l,n] (K=256) -------------
  __syncthreads();
  for (int ti = 0; ti < 2; ++ti) {
    int tile = wv * 2 + ti;
    int pr = tile >> 2, nc = tile & 3;
    fx8 sacc; for (int e = 0; e < 8; ++e) sacc[e] = 0.f;
    for (int kk = 0; kk < 256; kk += 32) {
      bfx16 afr, bfr;
      int p = pr * 16 + lr;
      #pragma unroll
      for (int j = 0; j < 8; ++j) {
        int la = kk + g0 + j, lb = kk + g1 + j;
        afr[j]     = tobf(fbf(sXd[la * 64 + p]) * sDec[la]);
        afr[8 + j] = tobf(fbf(sXd[lb * 64 + p]) * sDec[lb]);
      }
      int n = nc * 16 + lr;
      #pragma unroll
      for (int j = 0; j < 8; ++j) {
        bfr[j]     = sB[(kk + g0 + j) * 64 + n];
        bfr[8 + j] = sB[(kk + g1 + j) * 64 + n];
      }
      sacc = wmma_bf16(afr, bfr, sacc);
    }
    float* Sout = Stb + ((size_t)((b * NCH + c) * NH + hh)) * (HD * DST);
    int n = nc * 16 + lr;
    #pragma unroll
    for (int r = 0; r < 8; ++r) {
      int p = pr * 16 + r + hx * 8;
      Sout[(size_t)p * DST + n] = sacc[r];
    }
  }
}
constexpr size_t SSD1_SMEM = 3 * 256 * 64 * sizeof(bf16) + 2 * 256 * sizeof(float)
                             + 8 * 16 * 36 * sizeof(bf16);

// ===========================================================================
// SSD scan: sequential over 8 chunks per (b,h); stores prev-state in place.
// ===========================================================================
__global__ void __launch_bounds__(256) ssd_scan_kernel(
    float* __restrict__ Stb, const float* __restrict__ Csumb)
{
  int bh = blockIdx.x;
  int b = bh >> 4, hh = bh & 15;
  int t = threadIdx.x;
  float hst[16];
  for (int i = 0; i < 16; ++i) hst[i] = 0.f;
  for (int c = 0; c < NCH; ++c) {
    float dec = __expf(Csumb[((size_t)b * NCH + c) * NH + hh]);
    float* S = Stb + ((size_t)((b * NCH + c) * NH + hh)) * (HD * DST) + (size_t)t * 16;
    #pragma unroll
    for (int i = 0; i < 16; ++i) {
      float s = S[i];
      S[i] = hst[i];                 // prev-state for this chunk
      hst[i] = hst[i] * dec + s;
    }
  }
}

// ===========================================================================
// SSD part2: Y += exp(Acum) * (C @ prev^T) + D * x      (K = dstate = 64)
// ===========================================================================
__global__ void __launch_bounds__(256) ssd_part2_kernel(
    const float* __restrict__ XCb, const float* __restrict__ Acumb,
    const float* __restrict__ Stb, const float* __restrict__ Dp,
    float* __restrict__ Yb)
{
  int idx = blockIdx.x;
  int b = idx >> 7, c = (idx >> 4) & 7, hh = idx & 15;
  int t = threadIdx.x;
  int wv = t >> 5, lane = t & 31, hx = lane >> 4, lr = lane & 15;
  int g0 = hx * 8, g1 = 16 + hx * 8;
  size_t rowbase = (size_t)b * SEQ + (size_t)c * CHK;
  const float* St = Stb + ((size_t)((b * NCH + c) * NH + hh)) * (HD * DST);
  float dval = Dp[hh];

  for (int lt = 0; lt < 2; ++lt) {
    int l0 = wv * 32 + lt * 16;
    for (int pb = 0; pb < 4; ++pb) {
      fx8 acc; for (int e = 0; e < 8; ++e) acc[e] = 0.f;
      for (int kk = 0; kk < 64; kk += 32) {
        bfx16 afr, bfr;
        size_t arow = (rowbase + l0 + lr) * CONVD + DINNER + DST;
        #pragma unroll
        for (int j = 0; j < 8; ++j) {
          afr[j]     = tobf(XCb[arow + kk + g0 + j]);
          afr[8 + j] = tobf(XCb[arow + kk + g1 + j]);
        }
        int p = pb * 16 + lr;
        #pragma unroll
        for (int j = 0; j < 8; ++j) {
          bfr[j]     = tobf(St[(size_t)p * DST + kk + g0 + j]);
          bfr[8 + j] = tobf(St[(size_t)p * DST + kk + g1 + j]);
        }
        acc = wmma_bf16(afr, bfr, acc);
      }
      #pragma unroll
      for (int r = 0; r < 8; ++r) {
        int lrow = l0 + r + hx * 8;
        size_t grow = rowbase + lrow;
        float eg = __expf(Acumb[grow * NH + hh]);
        int p = pb * 16 + lr;
        size_t yi = grow * DINNER + (size_t)hh * HD + p;
        float xraw = XCb[grow * CONVD + hh * HD + p];
        Yb[yi] += acc[r] * eg + dval * xraw;
      }
    }
  }
}

// ===========================================================================
// y = rmsnorm(y * silu(z)) * gw      rows of 1024
// ===========================================================================
__global__ void __launch_bounds__(256) gate_rms_kernel(
    float* __restrict__ Yb, const float* __restrict__ Zb,
    const float* __restrict__ gw)
{
  __shared__ float buf[1024];
  __shared__ float red[256];
  int row = blockIdx.x, t = threadIdx.x;
  size_t base = (size_t)row * DINNER;
  float ss = 0.f;
  for (int c = t; c < DINNER; c += 256) {
    float v = Yb[base + c] * siluf(Zb[base + c]);
    buf[c] = v; ss += v * v;
  }
  red[t] = ss; __syncthreads();
  for (int o = 128; o > 0; o >>= 1) { if (t < o) red[t] += red[t + o]; __syncthreads(); }
  float inv = rsqrtf(red[0] / (float)DINNER + 1e-5f);
  for (int c = t; c < DINNER; c += 256) Yb[base + c] = buf[c] * inv * gw[c];
}

// ===========================================================================
// misc elementwise
// ===========================================================================
__global__ void flip_kernel(const float* __restrict__ in, float* __restrict__ out)
{
  size_t idx = (size_t)blockIdx.x * 256 + threadIdx.x;
  if (idx >= (size_t)MROWS * DMODEL) return;
  size_t row = idx / DMODEL; int c = (int)(idx % DMODEL);
  int tt = (int)(row & (SEQ - 1)); size_t brow = row - tt;
  out[idx] = in[(brow + (SEQ - 1 - tt)) * DMODEL + c];
}
__global__ void addflip_kernel(float* __restrict__ dst, const float* __restrict__ src)
{
  size_t idx = (size_t)blockIdx.x * 256 + threadIdx.x;
  if (idx >= (size_t)MROWS * DMODEL) return;
  size_t row = idx / DMODEL; int c = (int)(idx % DMODEL);
  int tt = (int)(row & (SEQ - 1)); size_t brow = row - tt;
  dst[idx] += src[(brow + (SEQ - 1 - tt)) * DMODEL + c];
}
__global__ void add_kernel(float* __restrict__ dst, const float* __restrict__ src, int n)
{
  int idx = blockIdx.x * 256 + threadIdx.x;
  if (idx < n) dst[idx] += src[idx];
}
__global__ void gelu_kernel(float* __restrict__ x, int n)
{
  int idx = blockIdx.x * 256 + threadIdx.x;
  if (idx < n) { float v = x[idx]; x[idx] = 0.5f * v * (1.f + erff(v * 0.70710678118f)); }
}
__global__ void __launch_bounds__(256) softmax_kernel(float* __restrict__ S, float scale)
{
  __shared__ float red[256];
  int row = blockIdx.x, t = threadIdx.x;
  float* r = S + (size_t)row * 1024;
  float m = -3.4e38f;
  for (int c = t; c < 1024; c += 256) m = fmaxf(m, r[c] * scale);
  red[t] = m; __syncthreads();
  for (int o = 128; o > 0; o >>= 1) { if (t < o) red[t] = fmaxf(red[t], red[t + o]); __syncthreads(); }
  float mx = red[0]; __syncthreads();
  float s = 0.f;
  for (int c = t; c < 1024; c += 256) { float e = __expf(r[c] * scale - mx); r[c] = e; s += e; }
  red[t] = s; __syncthreads();
  for (int o = 128; o > 0; o >>= 1) { if (t < o) red[t] += red[t + o]; __syncthreads(); }
  float inv = 1.f / red[0];
  for (int c = t; c < 1024; c += 256) r[c] *= inv;
}

// ===========================================================================
// host orchestration
// ===========================================================================
static inline void launch_gemm(const float* A, int lda, const float* W, int ldw,
                               int opW, const float* bias, float* C, int ldc,
                               int M, int N, int K, int accum, hipStream_t s)
{
  dim3 g((N + 127) / 128, M / 128);
  gemm_bf16_kernel<<<g, 256, 0, s>>>(A, lda, W, ldw, opW, bias, C, ldc, M, N, K, accum);
}

extern "C" void kernel_launch(void* const* d_in, const int* in_sizes, int n_in,
                              void* d_out, int out_size, void* d_ws, size_t ws_size,
                              hipStream_t stream)
{
  (void)in_sizes; (void)n_in; (void)out_size; (void)ws_size;
  float* ws = (float*)d_ws;

  const float* x_enc      = (const float*)d_in[0];
  const float* in_proj_w  = (const float*)d_in[2];
  const float* conv_w     = (const float*)d_in[3];
  const float* conv_b     = (const float*)d_in[4];
  const float* dt_bias    = (const float*)d_in[5];
  const float* A_log      = (const float*)d_in[6];
  const float* D_param    = (const float*)d_in[7];
  const float* gnorm_w    = (const float*)d_in[8];
  const float* out_proj_w = (const float*)d_in[9];
  const float* norm_w     = (const float*)d_in[10];
  const float* norm_b     = (const float*)d_in[11];
  const float* normf_w    = (const float*)d_in[12];
  const float* normf_b    = (const float*)d_in[13];
  const float* ca_nq_w    = (const float*)d_in[14];
  const float* ca_nq_b    = (const float*)d_in[15];
  const float* ca_nkv_w   = (const float*)d_in[16];
  const float* ca_nkv_b   = (const float*)d_in[17];
  const float* ca_qw      = (const float*)d_in[18];
  const float* ca_kw      = (const float*)d_in[19];
  const float* ca_vw      = (const float*)d_in[20];
  const float* ca_ow      = (const float*)d_in[21];
  const float* ca_qb      = (const float*)d_in[22];
  const float* ca_kb      = (const float*)d_in[23];
  const float* ca_vb      = (const float*)d_in[24];
  const float* ca_ob      = (const float*)d_in[25];
  const float* qp_w       = (const float*)d_in[26];
  const float* qp_b       = (const float*)d_in[27];
  const float* cp_w       = (const float*)d_in[28];
  const float* cp_b       = (const float*)d_in[29];
  const float* dec1_w     = (const float*)d_in[30];
  const float* dec1_b     = (const float*)d_in[31];
  const float* dec2_w     = (const float*)d_in[32];
  const float* dec2_b     = (const float*)d_in[33];

  const int EW  = (MROWS * DMODEL + 255) / 256;   // 512-wide elementwise grids

  auto mamba_dir = [&](const float* IN, int layer, int dir) {
    size_t ld = (size_t)(layer * 2 + dir);
    const float* Win  = in_proj_w + ld * DMODEL * DINPRJ;
    const float* cw   = conv_w    + ld * CONVD * 4;
    const float* cb   = conv_b    + ld * CONVD;
    const float* dtb  = dt_bias   + ld * NH;
    const float* Al   = A_log     + ld * NH;
    const float* Dp   = D_param   + ld * NH;
    const float* gw   = gnorm_w   + ld * DINNER;
    const float* Wout = out_proj_w + ld * DINNER * DMODEL;

    launch_gemm(IN, DMODEL, Win,                DINPRJ, 0, nullptr, ws + OZ,     DINNER, MROWS, DINNER, DMODEL, 0, stream);
    launch_gemm(IN, DMODEL, Win + DINNER,       DINPRJ, 0, nullptr, ws + OXBC,   CONVD,  MROWS, CONVD,  DMODEL, 0, stream);
    launch_gemm(IN, DMODEL, Win + DINNER+CONVD, DINPRJ, 0, nullptr, ws + ODTRAW, NH,     MROWS, NH,     DMODEL, 0, stream);
    dt_kernel<<<(MROWS * NH + 255) / 256, 256, 0, stream>>>(ws + ODTRAW, dtb, ws + ODT);
    conv_silu_kernel<<<(int)(((size_t)MROWS * CONVD + 255) / 256), 256, 0, stream>>>(
        ws + OXBC, cw, cb, ws + OXC);
    ssd_part1_kernel<<<BSZ * NCH * NH, 256, SSD1_SMEM, stream>>>(
        ws + OXC, ws + ODT, Al, ws + OY, ws + OST, ws + OACUM, ws + OCSUM);
    ssd_scan_kernel<<<BSZ * NH, 256, 0, stream>>>(ws + OST, ws + OCSUM);
    ssd_part2_kernel<<<BSZ * NCH * NH, 256, 0, stream>>>(
        ws + OXC, ws + OACUM, ws + OST, Dp, ws + OY);
    gate_rms_kernel<<<MROWS, 256, 0, stream>>>(ws + OY, ws + OZ, gw);
    launch_gemm(ws + OY, DINNER, Wout, DMODEL, 0, nullptr, ws + ODIR, DMODEL,
                MROWS, DMODEL, DINNER, 0, stream);
  };

  // ---- residual stream ----------------------------------------------------
  hipMemcpyAsync(ws + OH, x_enc, (size_t)MROWS * DMODEL * sizeof(float),
                 hipMemcpyDeviceToDevice, stream);
  for (int i = 0; i < 2; ++i) {
    if (i == 0)
      hipMemcpyAsync(ws + ORES, ws + OH, (size_t)MROWS * DMODEL * sizeof(float),
                     hipMemcpyDeviceToDevice, stream);
    else
      add_kernel<<<EW, 256, 0, stream>>>(ws + ORES, ws + OH, MROWS * DMODEL);
    ln_kernel<<<MROWS, 256, 0, stream>>>(ws + ORES, norm_w + i * DMODEL,
                                         norm_b + i * DMODEL, ws + OHN, DMODEL);
    mamba_dir(ws + OHN, i, 0);
    hipMemcpyAsync(ws + OH, ws + ODIR, (size_t)MROWS * DMODEL * sizeof(float),
                   hipMemcpyDeviceToDevice, stream);
    flip_kernel<<<EW, 256, 0, stream>>>(ws + OHN, ws + OFLIP);
    mamba_dir(ws + OFLIP, i, 1);
    addflip_kernel<<<EW, 256, 0, stream>>>(ws + OH, ws + ODIR);
  }
  add_kernel<<<EW, 256, 0, stream>>>(ws + ORES, ws + OH, MROWS * DMODEL);
  ln_kernel<<<MROWS, 256, 0, stream>>>(ws + ORES, normf_w, normf_b, ws + OHN, DMODEL);

  // ---- split into context / query (contiguous per batch) ------------------
  for (int b = 0; b < BSZ; ++b) {
    hipMemcpyAsync(ws + OCRAW + (size_t)b * 1024 * DMODEL,
                   ws + OHN + (size_t)b * SEQ * DMODEL,
                   (size_t)1024 * DMODEL * sizeof(float), hipMemcpyDeviceToDevice, stream);
    hipMemcpyAsync(ws + OQRAW + (size_t)b * 1024 * DMODEL,
                   ws + OHN + ((size_t)b * SEQ + 1024) * DMODEL,
                   (size_t)1024 * DMODEL * sizeof(float), hipMemcpyDeviceToDevice, stream);
  }
  launch_gemm(ws + OQRAW, DMODEL, qp_w, DMODEL, 0, qp_b, ws + OQIN, DMODEL, NQR, DMODEL, DMODEL, 0, stream);
  launch_gemm(ws + OCRAW, DMODEL, cp_w, DMODEL, 0, cp_b, ws + OCIN, DMODEL, NQR, DMODEL, DMODEL, 0, stream);
  ln_kernel<<<NQR, 256, 0, stream>>>(ws + OQIN, ca_nq_w,  ca_nq_b,  ws + OQN, DMODEL);
  ln_kernel<<<NQR, 256, 0, stream>>>(ws + OCIN, ca_nkv_w, ca_nkv_b, ws + OCN, DMODEL);
  launch_gemm(ws + OQN, DMODEL, ca_qw, DMODEL, 0, ca_qb, ws + OQM, DMODEL, NQR, DMODEL, DMODEL, 0, stream);
  launch_gemm(ws + OCN, DMODEL, ca_kw, DMODEL, 0, ca_kb, ws + OKM, DMODEL, NQR, DMODEL, DMODEL, 0, stream);
  launch_gemm(ws + OCN, DMODEL, ca_vw, DMODEL, 0, ca_vb, ws + OVM, DMODEL, NQR, DMODEL, DMODEL, 0, stream);

  // ---- attention: per (b,head) S = Q K^T ; softmax ; O = P V --------------
  for (int b = 0; b < BSZ; ++b)
    for (int h = 0; h < CAH; ++h) {
      const float* Qh = ws + OQM + ((size_t)b * 1024) * DMODEL + h * CAHD;
      const float* Kh = ws + OKM + ((size_t)b * 1024) * DMODEL + h * CAHD;
      float* Sh = ws + OS + ((size_t)(b * CAH + h)) * 1024 * 1024;
      launch_gemm(Qh, DMODEL, Kh, DMODEL, 1, nullptr, Sh, 1024, 1024, 1024, CAHD, 0, stream);
    }
  softmax_kernel<<<BSZ * CAH * 1024, 256, 0, stream>>>(ws + OS, 0.08838834764831845f);
  for (int b = 0; b < BSZ; ++b)
    for (int h = 0; h < CAH; ++h) {
      const float* Sh = ws + OS + ((size_t)(b * CAH + h)) * 1024 * 1024;
      const float* Vh = ws + OVM + ((size_t)b * 1024) * DMODEL + h * CAHD;
      float* Oh = ws + OOM + ((size_t)b * 1024) * DMODEL + h * CAHD;
      launch_gemm(Sh, 1024, Vh, DMODEL, 0, nullptr, Oh, DMODEL, 1024, CAHD, 1024, 0, stream);
    }
  // residual + output proj
  hipMemcpyAsync(ws + OATT, ws + OQIN, (size_t)NQR * DMODEL * sizeof(float),
                 hipMemcpyDeviceToDevice, stream);
  launch_gemm(ws + OOM, DMODEL, ca_ow, DMODEL, 0, ca_ob, ws + OATT, DMODEL,
              NQR, DMODEL, DMODEL, 1, stream);

  // ---- decoder MLP --------------------------------------------------------
  launch_gemm(ws + OATT, DMODEL, dec1_w, NHID, 0, dec1_b, ws + OD1, NHID,
              NQR, NHID, DMODEL, 0, stream);
  gelu_kernel<<<(NQR * NHID + 255) / 256, 256, 0, stream>>>(ws + OD1, NQR * NHID);
  launch_gemm(ws + OD1, NHID, dec2_w, NOUT, 0, dec2_b, (float*)d_out, NOUT,
              NQR, NOUT, NHID, 0, stream);
}
